// MultiHeadAttention_16922171147123
// MI455X (gfx1250) — compile-verified, matrix operand order checked by reading, not run on hardware
//
#include <hip/hip_runtime.h>

// MHA: B=8, S=1024, D=1024, H=16, DH=64. SCALE=0.5, LN_EPS=1e-5.
// d_out = [out f32 (8*1024*1024)] ++ [attn f32 (128*1024*1024)]
// Workspace layout (bytes):
//   qh  bf16 [128][1024][64]   @ 0        (16 MB)
//   kh  bf16 [128][1024][64]   @ 16 MB    (16 MB)
//   vt  bf16 [128][64][1024]   @ 32 MB    (16 MB)  (V transposed per head)
//   ctx f32  [8192][1024]      @ 48 MB    (32 MB)
//   proj f32 [8192][1024]      @ 80 MB    (32 MB)

typedef __attribute__((ext_vector_type(16))) __bf16         v16bf;
typedef __attribute__((ext_vector_type(8)))  float          v8f;
typedef __attribute__((ext_vector_type(4)))  float          v4f;
typedef __attribute__((ext_vector_type(8)))  unsigned short v8us;

union Frag {
    v16bf bf;
    v8us  us[2];
    unsigned short s[16];
};

__device__ __forceinline__ unsigned short f2bf(float f) {
    unsigned u = __float_as_uint(f);
    u += 0x7FFFu + ((u >> 16) & 1u);   // round-to-nearest-even
    return (unsigned short)(u >> 16);
}

// A/B fragment (16x32 / 32x16, bf16): per ISA layout, lane L<16 holds row (L&15),
// K = 0..7 and 16..23; lane L>=16 holds K = 8..15 and 24..31.
__device__ __forceinline__ void load_frag_lds(const unsigned short* ls, int rowBase,
                                              int lane, Frag& f) {
    const int row = rowBase + (lane & 15);
    const int kh  = (lane >> 4) << 3;             // 0 or 8
    const unsigned short* p = ls + row * 40 + kh; // 80B row stride, 16B aligned
    f.us[0] = *(const v8us*)(p);
    f.us[1] = *(const v8us*)(p + 16);
}

__device__ __forceinline__ void load_frag_g(const unsigned short* g, int ldg,
                                            int rowBase, int kBase, int lane, Frag& f) {
    const int row = rowBase + (lane & 15);
    const int kh  = kBase + ((lane >> 4) << 3);
    const unsigned short* p = g + (size_t)row * ldg + kh;
    f.us[0] = *(const v8us*)(p);
    f.us[1] = *(const v8us*)(p + 16);
}

// f32 source -> bf16 A-fragment (rows in `src` with stride lds32 floats)
__device__ __forceinline__ void load_frag_f32(const float* src, int lds32,
                                              int rowBase, int kBase, int lane, Frag& f) {
    const int row = rowBase + (lane & 15);
    const int kh  = kBase + ((lane >> 4) << 3);
    const float* p = src + (size_t)row * lds32 + kh;
    v4f f0 = *(const v4f*)(p),      f1 = *(const v4f*)(p + 4);
    v4f f2 = *(const v4f*)(p + 16), f3 = *(const v4f*)(p + 20);
    #pragma unroll
    for (int j = 0; j < 4; ++j) {
        f.s[j]      = f2bf(f0[j]);
        f.s[4 + j]  = f2bf(f1[j]);
        f.s[8 + j]  = f2bf(f2[j]);
        f.s[12 + j] = f2bf(f3[j]);
    }
}

__device__ __forceinline__ float wred_sum(float v) {
    #pragma unroll
    for (int o = 16; o > 0; o >>= 1) v += __shfl_xor(v, o, 32);
    return v;
}
__device__ __forceinline__ float wred_max(float v) {
    #pragma unroll
    for (int o = 16; o > 0; o >>= 1) v = fmaxf(v, __shfl_xor(v, o, 32));
    return v;
}

// ---------------------------------------------------------------------------
// C = A[f32 MxK] @ W^T (W f32 [N][K]) + bias.  M=8192, N=K=1024.
// MODE 0: bf16 head-major [BH][S][64]; MODE 2: bf16 transposed [BH][64][S];
// MODE 3: f32 row-major [M][N].  (compile-time -> straight-line epilogues)
// Block: 256 thr (8 waves), tile 128x128, BK=32. Wave = 32x64 (2x4 WMMA tiles).
// ---------------------------------------------------------------------------
template <int MODE>
__global__ __launch_bounds__(256) void gemm_xwT_kernel(
    const float* __restrict__ A, const float* __restrict__ W,
    const float* __restrict__ bias, void* __restrict__ dst) {
    constexpr int K = 1024;
    __shared__ unsigned short lsA[128 * 40];
    __shared__ unsigned short lsB[128 * 40];
    const int tid  = threadIdx.x;
    const int lane = tid & 31;
    const int wave = tid >> 5;
    const int wm = (wave & 3) * 32;
    const int wn = (wave >> 2) * 64;
    const int m0 = blockIdx.y * 128;
    const int n0 = blockIdx.x * 128;
    const int sr = tid >> 1;          // staging row 0..127
    const int sc = (tid & 1) * 16;    // staging col 0 or 16

    v8f acc[2][4] = {};

    for (int k0 = 0; k0 < K; k0 += 32) {
        __syncthreads();
        {   // stage A tile 128x32 f32 -> bf16
            const float* g = A + (size_t)(m0 + sr) * K + k0 + sc;
            unsigned short* d = &lsA[sr * 40 + sc];
            #pragma unroll
            for (int i = 0; i < 16; i += 4) {
                v4f f = *(const v4f*)(g + i);
                d[i + 0] = f2bf(f[0]); d[i + 1] = f2bf(f[1]);
                d[i + 2] = f2bf(f[2]); d[i + 3] = f2bf(f[3]);
            }
        }
        {   // stage W tile 128x32 f32 -> bf16 (rows of W are the N dimension)
            const float* g = W + (size_t)(n0 + sr) * K + k0 + sc;
            unsigned short* d = &lsB[sr * 40 + sc];
            #pragma unroll
            for (int i = 0; i < 16; i += 4) {
                v4f f = *(const v4f*)(g + i);
                d[i + 0] = f2bf(f[0]); d[i + 1] = f2bf(f[1]);
                d[i + 2] = f2bf(f[2]); d[i + 3] = f2bf(f[3]);
            }
        }
        __syncthreads();

        Frag fa[2], fb[4];
        #pragma unroll
        for (int mt = 0; mt < 2; ++mt) load_frag_lds(lsA, wm + mt * 16, lane, fa[mt]);
        #pragma unroll
        for (int nt = 0; nt < 4; ++nt) load_frag_lds(lsB, wn + nt * 16, lane, fb[nt]);
        #pragma unroll
        for (int mt = 0; mt < 2; ++mt)
            #pragma unroll
            for (int nt = 0; nt < 4; ++nt)
                acc[mt][nt] = __builtin_amdgcn_wmma_f32_16x16x32_bf16(
                    false, fa[mt].bf, false, fb[nt].bf, (short)0, acc[mt][nt],
                    false, false);
    }

    // Epilogue: C layout — lane 0-15: N=lane, M=r; lane 16-31: N=lane-16, M=8+r
    const int mrowBase = (lane >> 4) << 3;
    const int ncol = lane & 15;
    #pragma unroll
    for (int mt = 0; mt < 2; ++mt) {
        #pragma unroll
        for (int nt = 0; nt < 4; ++nt) {
            const int gn = n0 + wn + nt * 16 + ncol;
            const float bv = bias[gn];
            const int h = gn >> 6, dh = gn & 63;
            #pragma unroll
            for (int r = 0; r < 8; ++r) {
                const int gm = m0 + wm + mt * 16 + mrowBase + r;
                const float val = acc[mt][nt][r] + bv;
                if (MODE == 3) {
                    ((float*)dst)[(size_t)gm * 1024 + gn] = val;
                } else {
                    const int b = gm >> 10, s = gm & 1023;
                    unsigned short* o = (unsigned short*)dst;
                    if (MODE == 2)  // V transposed: [bh][dh][s]
                        o[(((size_t)(b * 16 + h)) * 64 + dh) * 1024 + s] = f2bf(val);
                    else            // Q/K: [bh][s][dh]
                        o[(((size_t)(b * 16 + h)) * 1024 + s) * 64 + dh] = f2bf(val);
                }
            }
        }
    }
}

// ---------------------------------------------------------------------------
// Fused attention: scores -> softmax -> attn write -> context, per 32-row slab.
// Block: 256 thr (8 waves), bh = blockIdx.z, rows m0..m0+31.
// LDS: f32 scores [32][1028] = 131584 B (dynamic). Row stride 1028 dwords:
// 16B-aligned, row*4 mod 64 banks distinct across 16 rows.
// Phase 1: wave w computes 32x128 strip (cols w*128..), K=64 from global bf16.
// Phase 2: wave w softmaxes rows 4w..4w+3; writes attn (only write) + LDS.
// Phase 3: wave w computes ctx tile (mt=w>>2, nt=w&3), K=1024, A from LDS f32.
// ---------------------------------------------------------------------------
#define LDSW 1028

__global__ __launch_bounds__(256) void attn_fused_kernel(
    const unsigned short* __restrict__ qh, const unsigned short* __restrict__ kh,
    const unsigned short* __restrict__ vt, float* __restrict__ attn,
    float* __restrict__ ctx) {
    extern __shared__ float scb[];
    const int tid = threadIdx.x, lane = tid & 31, wave = tid >> 5;
    const int bh = blockIdx.z;
    const int m0 = blockIdx.x * 32;
    const unsigned short* q = qh + (size_t)bh * 1024 * 64;
    const unsigned short* k = kh + (size_t)bh * 1024 * 64;
    const int rbase = (lane >> 4) << 3;
    const int ncol  = lane & 15;

    // ---- Phase 1: scores (x0.5) into LDS ----
    {
        Frag fa[2][2];
        #pragma unroll
        for (int mt = 0; mt < 2; ++mt)
            #pragma unroll
            for (int ks = 0; ks < 2; ++ks)
                load_frag_g(q, 64, m0 + mt * 16, ks * 32, lane, fa[mt][ks]);
        const int n0 = wave * 128;
        for (int nc = 0; nc < 8; ++nc) {
            const int nb = n0 + nc * 16;
            Frag fb0, fb1;
            load_frag_g(k, 64, nb, 0, lane, fb0);
            load_frag_g(k, 64, nb, 32, lane, fb1);
            v8f acc[2] = {};
            #pragma unroll
            for (int mt = 0; mt < 2; ++mt) {
                acc[mt] = __builtin_amdgcn_wmma_f32_16x16x32_bf16(
                    false, fa[mt][0].bf, false, fb0.bf, (short)0, acc[mt], false, false);
                acc[mt] = __builtin_amdgcn_wmma_f32_16x16x32_bf16(
                    false, fa[mt][1].bf, false, fb1.bf, (short)0, acc[mt], false, false);
            }
            const int col = nb + ncol;
            #pragma unroll
            for (int mt = 0; mt < 2; ++mt)
                #pragma unroll
                for (int r = 0; r < 8; ++r)
                    scb[(mt * 16 + rbase + r) * LDSW + col] = acc[mt][r] * 0.5f;
        }
    }
    __syncthreads();

    // ---- Phase 2: softmax rows, write attn (f32) + normalized back to LDS ----
    {
        float* aout = attn + (size_t)bh * 1024 * 1024;
        #pragma unroll
        for (int rr = 0; rr < 4; ++rr) {
            const int row = wave * 4 + rr;
            float* p = scb + row * LDSW;
            float x[32];
            float m = -3.4e38f;
            #pragma unroll
            for (int i = 0; i < 32; ++i) {
                x[i] = p[i * 32 + lane];
                m = fmaxf(m, x[i]);
            }
            m = wred_max(m);
            float s = 0.f;
            #pragma unroll
            for (int i = 0; i < 32; ++i) {
                x[i] = __expf(x[i] - m);
                s += x[i];
            }
            s = wred_sum(s);
            const float inv = 1.0f / s;
            float* g = aout + (size_t)(m0 + row) * 1024;
            #pragma unroll
            for (int i = 0; i < 32; ++i) {
                const float v = x[i] * inv;
                p[i * 32 + lane] = v;
                g[i * 32 + lane] = v;
            }
        }
    }
    __syncthreads();

    // ---- Phase 3: context tile per wave: [16x16], K=1024 ----
    {
        const unsigned short* v = vt + (size_t)bh * 64 * 1024;
        const int mt = wave >> 2;      // 0..1
        const int nt = wave & 3;       // 0..3
        v8f cacc = {};
        for (int k0 = 0; k0 < 1024; k0 += 32) {
            Frag fav, fbv;
            load_frag_f32(scb, LDSW, mt * 16, k0, lane, fav);
            load_frag_g(v, 1024, nt * 16, k0, lane, fbv);
            cacc = __builtin_amdgcn_wmma_f32_16x16x32_bf16(
                false, fav.bf, false, fbv.bf, (short)0, cacc, false, false);
        }
        const int b = bh >> 4, h = bh & 15;
        const int dh = nt * 16 + ncol;
        #pragma unroll
        for (int r = 0; r < 8; ++r) {
            const int srow = m0 + mt * 16 + rbase + r;
            ctx[((size_t)(b * 1024 + srow)) * 1024 + h * 64 + dh] = cacc[r];
        }
    }
}

// ---------------------------------------------------------------------------
// out = LayerNorm(residual + proj) * gamma + beta, per row of 1024.
// ---------------------------------------------------------------------------
__global__ __launch_bounds__(256) void ln_kernel(
    const float* __restrict__ proj, const float* __restrict__ residual,
    const float* __restrict__ gamma, const float* __restrict__ beta,
    float* __restrict__ out) {
    __shared__ float ssum[8];
    __shared__ float ssum2[8];
    const size_t row = blockIdx.x;
    const int tid = threadIdx.x, lane = tid & 31, wave = tid >> 5;
    v4f y = *(const v4f*)(proj + row * 1024 + tid * 4);
    v4f rz = *(const v4f*)(residual + row * 1024 + tid * 4);
    float s = 0.f, s2 = 0.f;
    #pragma unroll
    for (int i = 0; i < 4; ++i) {
        y[i] += rz[i];
        s += y[i];
        s2 += y[i] * y[i];
    }
    s = wred_sum(s);
    s2 = wred_sum(s2);
    if (lane == 0) { ssum[wave] = s; ssum2[wave] = s2; }
    __syncthreads();
    s = 0.f; s2 = 0.f;
    #pragma unroll
    for (int i = 0; i < 8; ++i) { s += ssum[i]; s2 += ssum2[i]; }
    const float mu = s * (1.0f / 1024.0f);
    const float var = s2 * (1.0f / 1024.0f) - mu * mu;
    const float inv = rsqrtf(var + 1e-5f);
    v4f g = *(const v4f*)(gamma + tid * 4);
    v4f be = *(const v4f*)(beta + tid * 4);
    #pragma unroll
    for (int i = 0; i < 4; ++i) y[i] = (y[i] - mu) * inv * g[i] + be[i];
    *(v4f*)(out + row * 1024 + tid * 4) = y;
}

extern "C" void kernel_launch(void* const* d_in, const int* in_sizes, int n_in,
                              void* d_out, int out_size, void* d_ws, size_t ws_size,
                              hipStream_t stream) {
    const float* key   = (const float*)d_in[0];
    const float* value = (const float*)d_in[1];
    const float* query = (const float*)d_in[2];
    const float* Wq = (const float*)d_in[3];
    const float* bq = (const float*)d_in[4];
    const float* Wk = (const float*)d_in[5];
    const float* bk = (const float*)d_in[6];
    const float* Wv = (const float*)d_in[7];
    const float* bv = (const float*)d_in[8];
    const float* Wo = (const float*)d_in[9];
    const float* bo = (const float*)d_in[10];
    const float* gamma = (const float*)d_in[11];
    const float* beta  = (const float*)d_in[12];

    float* out  = (float*)d_out;
    float* attn = out + (size_t)8 * 1024 * 1024;

    char* ws = (char*)d_ws;
    unsigned short* qh = (unsigned short*)(ws);
    unsigned short* kh = (unsigned short*)(ws + (16ull << 20));
    unsigned short* vt = (unsigned short*)(ws + (32ull << 20));
    float* ctx  = (float*)(ws + (48ull << 20));
    float* proj = (float*)(ws + (80ull << 20));

    const dim3 gg(8, 64, 1);  // N/128, M/128
    gemm_xwT_kernel<0><<<gg, 256, 0, stream>>>(query, Wq, bq, qh);
    gemm_xwT_kernel<0><<<gg, 256, 0, stream>>>(key,   Wk, bk, kh);
    gemm_xwT_kernel<2><<<gg, 256, 0, stream>>>(value, Wv, bv, vt);

    const size_t lds_bytes = (size_t)32 * LDSW * sizeof(float);  // 131584 B
    attn_fused_kernel<<<dim3(32, 1, 128), 256, lds_bytes, stream>>>(qh, kh, vt, attn, ctx);

    gemm_xwT_kernel<3><<<gg, 256, 0, stream>>>(ctx, Wo, bo, proj);
    ln_kernel<<<dim3(8192), 256, 0, stream>>>(proj, query, gamma, beta, out);
}
